// ConvGNN_43379169689791
// MI455X (gfx1250) — matrix operand order repaired and checked by reading, not compiled, not run on hardware
//
#include <hip/hip_runtime.h>

// ---------------------------------------------------------------------------
// ConvGNN (2-layer GCN) for MI455X / gfx1250.
//   agg = scatter_add_e( x[src]*norm_out[src]*norm_in[dst] )   (norms fused)
//   h   = relu(agg @ W1 + b1)          -> kept in bf16 (L2-resident, half bytes)
//   out = (scatter over h) @ W2 + b2 + x
// GEMMs run on v_wmma_f32_16x16x32_bf16, one 16x16 tile per wave32,
// fragments loaded as contiguous b128 global loads per the ISA VGPR layouts.
// ---------------------------------------------------------------------------

typedef __attribute__((ext_vector_type(16))) __bf16       v16bf;
typedef __attribute__((ext_vector_type(8)))  float        v8f;
typedef __attribute__((ext_vector_type(4)))  unsigned int v4u;

union FragCvt { v4u q[2]; v16bf v; };

__device__ __forceinline__ unsigned short f32_to_bf16(float f) {
  union { float f; unsigned int u; } c; c.f = f;
  unsigned int u = c.u;
  unsigned int r = u + 0x7FFFu + ((u >> 16) & 1u);   // round-to-nearest-even
  return (unsigned short)(r >> 16);
}

__device__ __forceinline__ float bf16_to_f32(unsigned int h) {
  union { unsigned int u; float f; } c; c.u = h << 16; return c.f;
}

// ---------------------------------------------------------------------------
// Degree accumulation: one thread per edge, float atomics (L2-resident).
__global__ __launch_bounds__(256) void degree_kernel(
    const long long* __restrict__ src, const long long* __restrict__ dst,
    float* __restrict__ deg_out, float* __restrict__ deg_in, int E) {
  int e = blockIdx.x * blockDim.x + threadIdx.x;
  if (e >= E) return;
  atomicAdd(deg_out + (int)src[e], 1.0f);
  atomicAdd(deg_in  + (int)dst[e], 1.0f);
}

// In-place deg -> rsqrt(max(deg,1)). Covers both norm arrays (contiguous, 2N).
__global__ __launch_bounds__(256) void norm_kernel(float* __restrict__ deg, int n) {
  int i = blockIdx.x * blockDim.x + threadIdx.x;
  if (i >= n) return;
  float d = deg[i];
  d = d < 1.0f ? 1.0f : d;
  deg[i] = rsqrtf(d);
}

// W [K, Nout] f32 row-major  ->  WT [Nout, K] bf16 (so B-fragments load like A)
__global__ __launch_bounds__(256) void wtranspose_kernel(
    const float* __restrict__ W, unsigned short* __restrict__ WT, int K, int Nout) {
  int idx = blockIdx.x * blockDim.x + threadIdx.x;
  if (idx >= K * Nout) return;
  int k = idx / Nout, n = idx % Nout;
  WT[n * K + k] = f32_to_bf16(W[idx]);
}

// Layer-1 scatter: one wave per edge, 4 f32 per lane (128 cols), norms fused.
__global__ __launch_bounds__(256) void scatter_f32_kernel(
    const float* __restrict__ x,
    const long long* __restrict__ src, const long long* __restrict__ dst,
    const float* __restrict__ norm_out, const float* __restrict__ norm_in,
    float* __restrict__ agg, int E) {
  int t = blockIdx.x * blockDim.x + threadIdx.x;
  int e = t >> 5, lane = t & 31;
  if (e >= E) return;
  int s = (int)src[e], d = (int)dst[e];
  float sc = norm_out[s] * norm_in[d];
  float4 v = *((const float4*)(x + (size_t)s * 128) + lane);
  float* ar = agg + (size_t)d * 128 + lane * 4;
  atomicAdd(ar + 0, v.x * sc);
  atomicAdd(ar + 1, v.y * sc);
  atomicAdd(ar + 2, v.z * sc);
  atomicAdd(ar + 3, v.w * sc);
}

// Layer-2 scatter: one wave per edge, 8 bf16 per lane (256 cols) from h_bf.
__global__ __launch_bounds__(256) void scatter_bf16_kernel(
    const unsigned short* __restrict__ h,
    const long long* __restrict__ src, const long long* __restrict__ dst,
    const float* __restrict__ norm_out, const float* __restrict__ norm_in,
    float* __restrict__ agg, int E) {
  int t = blockIdx.x * blockDim.x + threadIdx.x;
  int e = t >> 5, lane = t & 31;
  if (e >= E) return;
  int s = (int)src[e], d = (int)dst[e];
  float sc = norm_out[s] * norm_in[d];
  v4u q = *(const v4u*)(h + (size_t)s * 256 + lane * 8);
  float* ar = agg + (size_t)d * 256 + lane * 8;
#pragma unroll
  for (int j = 0; j < 4; ++j) {
    atomicAdd(ar + 2 * j + 0, bf16_to_f32(q[j] & 0xFFFFu) * sc);
    atomicAdd(ar + 2 * j + 1, bf16_to_f32(q[j] >> 16) * sc);
  }
}

// Elementwise f32 -> bf16.
__global__ __launch_bounds__(256) void f2bf_kernel(
    const float* __restrict__ in, unsigned short* __restrict__ out, int n) {
  int i = blockIdx.x * blockDim.x + threadIdx.x;
  if (i >= n) return;
  out[i] = f32_to_bf16(in[i]);
}

// ---------------------------------------------------------------------------
// WMMA GEMM: D[M,NOUT] = A[M,K](bf16) * BT[NOUT,K](bf16)^T + epilogue.
// One 16x16 tile per wave; 8 waves per block. K stepped by 32.
// MODE 0: relu(acc + bias) -> bf16 out.   MODE 1: acc + bias + resid -> f32 out.
template <int K, int NOUT, int MODE>
__global__ __launch_bounds__(256) void gemm_wmma_kernel(
    const unsigned short* __restrict__ A,
    const unsigned short* __restrict__ BT,
    const float* __restrict__ bias,
    const float* __restrict__ resid,
    void* __restrict__ outp, int M) {
  const int lane = threadIdx.x & 31;
  const int wave = threadIdx.x >> 5;
  const int tilesN = NOUT / 16;
  const int tilesM = M / 16;
  const int tile = blockIdx.x * 8 + wave;           // wave-uniform
  if (tile >= tilesM * tilesN) return;              // whole wave exits together
  const int mt = tile / tilesN;
  const int nt = tile % tilesN;

  const int r  = lane & 15;                         // A-row / B-col / D-col
  const int hi = lane >> 4;                         // lane half
  const size_t aRow = (size_t)(mt * 16 + r) * K;
  const size_t bRow = (size_t)(nt * 16 + r) * K;

  v8f acc = {};
#pragma unroll
  for (int k0 = 0; k0 < K; k0 += 32) {
    FragCvt fa, fb;
    // A 16-bit 16x32 layout: lanes>=16 hold K+8; VGPRs 4-7 hold K+16.
    const int ka = k0 + hi * 8;
    fa.q[0] = *(const v4u*)(A + aRow + ka);
    fa.q[1] = *(const v4u*)(A + aRow + ka + 16);
    // B 16-bit 32x16 layout (BT row-major): lanes>=16 hold K+8; VGPRs 4-7 hold K+16.
    const int kb = k0 + hi * 8;
    fb.q[0] = *(const v4u*)(BT + bRow + kb);
    fb.q[1] = *(const v4u*)(BT + bRow + kb + 16);
    acc = __builtin_amdgcn_wmma_f32_16x16x32_bf16(
        /*neg_a=*/false, fa.v, /*neg_b=*/false, fb.v,
        /*c_mod=*/(short)0, acc, /*reuse_a=*/false, /*reuse_b=*/false);
  }

  // C/D layout: lane half hi -> M = i + 8*hi; N = lane%16.
  const int nCol = nt * 16 + r;
  const int mBase = mt * 16 + hi * 8;
  const float bv = bias[nCol];
#pragma unroll
  for (int i = 0; i < 8; ++i) {
    const size_t idx = (size_t)(mBase + i) * NOUT + nCol;
    float v = acc[i] + bv;
    if (MODE == 0) {
      v = v > 0.0f ? v : 0.0f;
      ((unsigned short*)outp)[idx] = f32_to_bf16(v);
    } else {
      ((float*)outp)[idx] = v + resid[idx];
    }
  }
}

// ---------------------------------------------------------------------------
extern "C" void kernel_launch(void* const* d_in, const int* in_sizes, int n_in,
                              void* d_out, int out_size, void* d_ws, size_t ws_size,
                              hipStream_t stream) {
  (void)in_sizes; (void)n_in; (void)out_size; (void)ws_size;
  constexpr int N  = 50000;
  constexpr int E  = 800000;
  constexpr int F1 = 128;
  constexpr int F2 = 256;

  const float*     x   = (const float*)d_in[0];      // [N, F1]
  const float*     W1  = (const float*)d_in[1];      // [F1, F2]
  const float*     b1  = (const float*)d_in[2];      // [F2]
  const float*     W2  = (const float*)d_in[3];      // [F2, F1]
  const float*     b2  = (const float*)d_in[4];      // [F1]
  const long long* src = (const long long*)d_in[5];  // [E] int64
  const long long* dst = (const long long*)d_in[6];  // [E] int64
  float* out = (float*)d_out;                        // [N, F1]

  // Workspace carve-out (256B aligned).
  char* ws = (char*)d_ws;
  size_t off = 0;
  auto carve = [&](size_t bytes) -> char* {
    char* p = ws + off;
    off += (bytes + 255) & ~(size_t)255;
    return p;
  };
  float*          norms = (float*)carve((size_t)2 * N * sizeof(float)); // out|in
  float*          norm_out = norms;
  float*          norm_in  = norms + N;
  float*          agg  = (float*)carve((size_t)N * F2 * sizeof(float));
  unsigned short* Abf  = (unsigned short*)carve((size_t)N * F2 * 2);
  unsigned short* h_bf = (unsigned short*)carve((size_t)N * F2 * 2);
  unsigned short* W1T  = (unsigned short*)carve((size_t)F2 * F1 * 2);
  unsigned short* W2T  = (unsigned short*)carve((size_t)F1 * F2 * 2);

  const int TB = 256;

  // Degree norms (shared by both layers).
  hipMemsetAsync(norms, 0, (size_t)2 * N * sizeof(float), stream);
  degree_kernel<<<(E + TB - 1) / TB, TB, 0, stream>>>(src, dst, norm_out, norm_in, E);
  norm_kernel<<<(2 * N + TB - 1) / TB, TB, 0, stream>>>(norms, 2 * N);

  // Weights -> transposed bf16 (tiny).
  wtranspose_kernel<<<(F1 * F2 + TB - 1) / TB, TB, 0, stream>>>(W1, W1T, F1, F2);
  wtranspose_kernel<<<(F1 * F2 + TB - 1) / TB, TB, 0, stream>>>(W2, W2T, F2, F1);

  // ---- Layer 1: scatter(x, fused norms) -> agg[N,F1]; bf16; GEMM -> h_bf.
  hipMemsetAsync(agg, 0, (size_t)N * F1 * sizeof(float), stream);
  scatter_f32_kernel<<<(E * 32) / TB, TB, 0, stream>>>(x, src, dst, norm_out,
                                                       norm_in, agg, E);
  f2bf_kernel<<<(N * F1 + TB - 1) / TB, TB, 0, stream>>>(agg, Abf, N * F1);
  {
    const int tiles = (N / 16) * (F2 / 16);          // 50000, /8 = 6250 blocks
    gemm_wmma_kernel<F1, F2, 0><<<tiles / 8, TB, 0, stream>>>(
        Abf, W1T, b1, nullptr, (void*)h_bf, N);
  }

  // ---- Layer 2: scatter(h_bf) -> agg[N,F2]; bf16; GEMM + residual -> out.
  hipMemsetAsync(agg, 0, (size_t)N * F2 * sizeof(float), stream);
  scatter_bf16_kernel<<<(E * 32) / TB, TB, 0, stream>>>(h_bf, src, dst, norm_out,
                                                        norm_in, agg, E);
  f2bf_kernel<<<(N * F2 + TB - 1) / TB, TB, 0, stream>>>(agg, Abf, N * F2);
  {
    const int tiles = (N / 16) * (F1 / 16);          // 25000, /8 = 3125 blocks
    gemm_wmma_kernel<F2, F1, 1><<<tiles / 8, TB, 0, stream>>>(
        Abf, W2T, b2, x, (void*)out, N);
  }
}
